// DeformableBasicBlock_35759897706918
// MI455X (gfx1250) — matrix operand order repaired and checked by reading, not compiled, not run on hardware
//
#include <hip/hip_runtime.h>
#include <hip/hip_bf16.h>

typedef __attribute__((ext_vector_type(16))) _Float16 v16h;
typedef __attribute__((ext_vector_type(8)))  _Float16 v8h;
typedef __attribute__((ext_vector_type(8)))  float    v8f;

#define CCH 128
#define HH  128
#define WW  128
#define KTAP 9
#define KDIM (CCH * KTAP)     // 1152, K reordered tap-major: k = tap*128 + cin
#define KSTEP 32
#define NIT (KDIM / KSTEP)    // 36 K-iterations; each chunk has a single tap
#define MT 64                 // pixels per workgroup (64-wide row segment)
#define NSUB 4                // 16-pixel m-subtiles per wave

// ---------------------------------------------------------------------------
// Kernel 0: transpose+convert weights to f16 tap-major  wT[n][tap*128+cin]
// ---------------------------------------------------------------------------
__global__ __launch_bounds__(256) void k0_prep_weights(
    const float* __restrict__ w1, const float* __restrict__ w2,
    _Float16* __restrict__ w1T, _Float16* __restrict__ w2T)
{
    const int idx = blockIdx.x * 256 + threadIdx.x;
    if (idx < CCH * KDIM) {
        const int n   = idx / KDIM;
        const int r   = idx - n * KDIM;
        const int tap = r >> 7;
        const int cin = r & 127;
        const int src = n * KDIM + cin * KTAP + tap;
        w1T[idx] = (_Float16)w1[src];
        w2T[idx] = (_Float16)w2[src];
    }
}

// ---------------------------------------------------------------------------
// Kernel 1: deformable conv as f16 WMMA GEMM (tap-major K), BN1+ReLU -> f16 mid
// ---------------------------------------------------------------------------
__global__ __launch_bounds__(256) void k1_deform_gemm(
    const float* __restrict__ x, const float* __restrict__ off,
    const _Float16* __restrict__ w1T,
    const float* __restrict__ g1, const float* __restrict__ b1,
    const float* __restrict__ mu1, const float* __restrict__ var1,
    _Float16* __restrict__ mid)
{
    __shared__ int4     sO[MT * KTAP];       // 4 clamped corner offsets
    __shared__ float4   sW[MT * KTAP];       // 4 corner weights (zeroed if OOB)
    __shared__ _Float16 Als[2][MT * KSTEP];  // double-buffered A tile (64x32)

    const int tid  = threadIdx.x;
    const int lane = tid & 31;
    const int wave = tid >> 5;

    const int pix0 = blockIdx.x * MT;
    const int b    = pix0 >> 14;
    const int rem  = pix0 & 16383;
    const int h    = rem >> 7;
    const int w0   = rem & 127;              // multiple of 64

    // --- precompute bilinear corners/weights for 64 pixels x 9 taps ---
    for (int t = tid; t < MT * KTAP; t += 256) {
        const int p = t / KTAP, tap = t - p * KTAP;
        const int w = w0 + p;
        const float dy = off[(((size_t)b * 18 + 2 * tap    ) * HH + h) * WW + w];
        const float dx = off[(((size_t)b * 18 + 2 * tap + 1) * HH + h) * WW + w];
        const float py = (float)h + (float)(tap / 3 - 1) + dy;
        const float px = (float)w + (float)(tap % 3 - 1) + dx;
        const float y0f = floorf(py), x0f = floorf(px);
        const int   y0 = (int)y0f,   x0 = (int)x0f;
        const float fy = py - y0f,   fx = px - x0f;
        const float ey = 1.f - fy,   ex = 1.f - fx;
        const bool vy0 = (unsigned)y0       < HH, vy1 = (unsigned)(y0 + 1) < HH;
        const bool vx0 = (unsigned)x0       < WW, vx1 = (unsigned)(x0 + 1) < WW;
        const int y0c = min(max(y0, 0), HH - 1), y1c = min(max(y0 + 1, 0), HH - 1);
        const int x0c = min(max(x0, 0), WW - 1), x1c = min(max(x0 + 1, 0), WW - 1);
        sO[t] = int4{(y0c << 7) + x0c, (y0c << 7) + x1c,
                     (y1c << 7) + x0c, (y1c << 7) + x1c};
        sW[t] = float4{(vy0 && vx0) ? ey * ex : 0.f, (vy0 && vx1) ? ey * fx : 0.f,
                       (vy1 && vx0) ? fy * ex : 0.f, (vy1 && vx1) ? fy * fx : 0.f};
    }
    __syncthreads();

    // per-lane BN constants (each lane owns one output channel)
    const int   nglob = wave * 16 + (lane & 15);
    const float inv   = g1[nglob] * rsqrtf(var1[nglob] + 1e-5f);
    const float add   = b1[nglob] - mu1[nglob] * inv;

    const int half = lane >> 4;
    const int mrow = lane & 15;
    const int pgrp = tid & 63;    // pixel this thread gathers
    const int cgrp = tid >> 6;    // which 8-cin group (0..3)

    v8f acc[NSUB] = {};

    for (int it = 0; it < NIT; ++it) {
        const int k0  = it * KSTEP;
        const int tap = k0 >> 7;            // single tap per chunk (tap-major K)
        const int c0  = k0 & 127;
        const int buf = it & 1;

        // gather 8 consecutive cin for one (pixel, tap); coords read ONCE
        {
            const int4   o = sO[pgrp * KTAP + tap];
            const float4 g = sW[pgrp * KTAP + tap];
            const float* xb = x + ((size_t)(b * CCH + c0 + cgrp * 8) << 14);
            v8h tmp;
            #pragma unroll
            for (int j = 0; j < 8; ++j) {
                const float v = g.x * xb[o.x] + g.y * xb[o.y]
                              + g.z * xb[o.z] + g.w * xb[o.w];
                tmp[j] = (_Float16)v;
                xb += (1 << 14);            // next cin plane: +64KB immediate
            }
            *(v8h*)&Als[buf][(pgrp << 5) + (cgrp << 3)] = tmp;
        }
        __syncthreads();

        // B fragment directly from global (w1T is 294KB, L2-hot)
        v16h bv;
        {
            const _Float16* pb = w1T + (size_t)nglob * KDIM + k0 + half * 8;
            #pragma unroll
            for (int j = 0; j < 8; ++j) { bv[j] = pb[j]; bv[j + 8] = pb[16 + j]; }
        }
        // 4 m-subtiles, back-to-back WMMAs sharing bv
        #pragma unroll
        for (int s = 0; s < NSUB; ++s) {
            v16h av;
            const _Float16* pa = &Als[buf][((s * 16 + mrow) << 5) + half * 8];
            #pragma unroll
            for (int j = 0; j < 8; ++j) { av[j] = pa[j]; av[j + 8] = pa[16 + j]; }
            acc[s] = __builtin_amdgcn_wmma_f32_16x16x32_f16(
                         false, av, false, bv, (short)0, acc[s], false, false);
        }
    }

    // epilogue: BN1 + ReLU -> f16 intermediate  mid[b, n, h, w0 + pixel]
    _Float16* mb = mid + (((size_t)(b * CCH + nglob) << 14) + (h << 7) + w0);
    #pragma unroll
    for (int s = 0; s < NSUB; ++s) {
        #pragma unroll
        for (int r = 0; r < 8; ++r) {
            const int mo = s * 16 + r + (half << 3);
            float v = acc[s][r] * inv + add;
            v = v > 0.f ? v : 0.f;
            mb[mo] = (_Float16)v;
        }
    }
}

// ---------------------------------------------------------------------------
// Kernel 2: standard 3x3 conv (zero pad) WMMA GEMM, BN2 + residual + ReLU
// ---------------------------------------------------------------------------
__global__ __launch_bounds__(256) void k2_conv_gemm(
    const _Float16* __restrict__ mid, const _Float16* __restrict__ w2T,
    const float* __restrict__ g2, const float* __restrict__ b2,
    const float* __restrict__ mu2, const float* __restrict__ var2,
    const float* __restrict__ x, float* __restrict__ out)
{
    __shared__ _Float16 Als[2][MT * KSTEP];

    const int tid  = threadIdx.x;
    const int lane = tid & 31;
    const int wave = tid >> 5;

    const int pix0 = blockIdx.x * MT;
    const int b    = pix0 >> 14;
    const int rem  = pix0 & 16383;
    const int h    = rem >> 7;
    const int w0   = rem & 127;

    const int   nglob = wave * 16 + (lane & 15);
    const float inv   = g2[nglob] * rsqrtf(var2[nglob] + 1e-5f);
    const float add   = b2[nglob] - mu2[nglob] * inv;

    const int half = lane >> 4;
    const int mrow = lane & 15;
    const int pgrp = tid & 63;
    const int cgrp = tid >> 6;

    v8f acc[NSUB] = {};

    for (int it = 0; it < NIT; ++it) {
        const int k0  = it * KSTEP;
        const int tap = k0 >> 7;
        const int c0  = k0 & 127;
        const int buf = it & 1;

        // im2col: single bounds test per (pixel, tap), then 8 strided f16 loads
        {
            const int yy = h + (tap / 3) - 1;
            const int xx = w0 + pgrp + (tap % 3) - 1;
            const bool valid = ((unsigned)yy < HH) && ((unsigned)xx < WW);
            const _Float16* mb =
                mid + ((size_t)(b * CCH + c0 + cgrp * 8) << 14) + (yy << 7) + xx;
            v8h tmp;
            #pragma unroll
            for (int j = 0; j < 8; ++j)
                tmp[j] = valid ? mb[(size_t)j << 14] : (_Float16)0.f;
            *(v8h*)&Als[buf][(pgrp << 5) + (cgrp << 3)] = tmp;
        }
        __syncthreads();

        v16h bv;
        {
            const _Float16* pb = w2T + (size_t)nglob * KDIM + k0 + half * 8;
            #pragma unroll
            for (int j = 0; j < 8; ++j) { bv[j] = pb[j]; bv[j + 8] = pb[16 + j]; }
        }
        #pragma unroll
        for (int s = 0; s < NSUB; ++s) {
            v16h av;
            const _Float16* pa = &Als[buf][((s * 16 + mrow) << 5) + half * 8];
            #pragma unroll
            for (int j = 0; j < 8; ++j) { av[j] = pa[j]; av[j + 8] = pa[16 + j]; }
            acc[s] = __builtin_amdgcn_wmma_f32_16x16x32_f16(
                         false, av, false, bv, (short)0, acc[s], false, false);
        }
    }

    // epilogue: BN2 + identity residual + ReLU -> fp32 output
    const size_t chanBase = ((size_t)(b * CCH + nglob) << 14) + (h << 7) + w0;
    const float* xb = x + chanBase;
    float*       ob = out + chanBase;
    #pragma unroll
    for (int s = 0; s < NSUB; ++s) {
        #pragma unroll
        for (int r = 0; r < 8; ++r) {
            const int mo = s * 16 + r + (half << 3);
            float v = acc[s][r] * inv + add + xb[mo];
            v = v > 0.f ? v : 0.f;
            ob[mo] = v;
        }
    }
}

// ---------------------------------------------------------------------------
extern "C" void kernel_launch(void* const* d_in, const int* in_sizes, int n_in,
                              void* d_out, int out_size, void* d_ws, size_t ws_size,
                              hipStream_t stream) {
    const float* x    = (const float*)d_in[0];
    const float* off  = (const float*)d_in[1];
    const float* w1   = (const float*)d_in[2];
    const float* g1   = (const float*)d_in[3];
    const float* b1   = (const float*)d_in[4];
    const float* mu1  = (const float*)d_in[5];
    const float* var1 = (const float*)d_in[6];
    const float* w2   = (const float*)d_in[7];
    const float* g2   = (const float*)d_in[8];
    const float* b2   = (const float*)d_in[9];
    const float* mu2  = (const float*)d_in[10];
    const float* var2 = (const float*)d_in[11];

    float* out = (float*)d_out;

    char* ws = (char*)d_ws;
    _Float16* mid = (_Float16*)ws;                                   // 16.78 MB
    _Float16* w1T = (_Float16*)(ws + (size_t)4 * CCH * HH * WW * 2); // 294 KB
    _Float16* w2T = w1T + (size_t)CCH * KDIM;                        // 294 KB

    const int npix   = 4 * HH * WW;   // 65536
    const int blocks = npix / MT;     // 1024

    k0_prep_weights<<<(CCH * KDIM + 255) / 256, 256, 0, stream>>>(w1, w2, w1T, w2T);
    k1_deform_gemm<<<blocks, 256, 0, stream>>>(x, off, w1T, g1, b1, mu1, var1, mid);
    k2_conv_gemm  <<<blocks, 256, 0, stream>>>(mid, w2T, g2, b2, mu2, var2, x, out);
}
